// BipartiteGNN_69904887709993
// MI455X (gfx1250) — matrix operand order repaired and checked by reading, not compiled, not run on hardware
//
#include <hip/hip_runtime.h>
#include <hip/hip_bf16.h>

// ---------------- problem constants ----------------
#define NU_   200000
#define NP_   50000
#define EED_  500000
#define DU_   100
#define DP_   50
#define HDIM  128
#define NTOT  (NU_ + NP_)
#define BN_EPS 1e-5f

// ---------------- WMMA vector types ----------------
typedef __attribute__((ext_vector_type(16))) _Float16 v16h;
typedef __attribute__((ext_vector_type(8)))  _Float16 h8;
typedef __attribute__((ext_vector_type(8)))  float    v8f;
typedef int v4i_ __attribute__((vector_size(16)));     // matches builtin param type

#define BM    128          // rows per block
#define KSTR  136          // LDS row stride (halves): 272B rows -> conflict-free, 16B aligned

// ---------------- gfx1250 async global->LDS copy (ASYNCcnt) ----------------
__device__ __forceinline__ void async_copy_b128(const void* g, void* l) {
#if __has_builtin(__builtin_amdgcn_global_load_async_to_lds_b128)
    __builtin_amdgcn_global_load_async_to_lds_b128(
        (__attribute__((address_space(1))) v4i_*)g,
        (__attribute__((address_space(3))) v4i_*)l, 0, 0);
#else
    *(h8*)l = *(const h8*)g;   // fallback: VGPR round-trip
#endif
}
__device__ __forceinline__ void async_wait_all() {
#if __has_builtin(__builtin_amdgcn_s_wait_asynccnt)
    __builtin_amdgcn_s_wait_asynccnt(0);
#elif defined(__gfx1250__)
    asm volatile("s_wait_asynccnt 0" ::: "memory");
#endif
}

// =====================================================================
// Shared WMMA compute core: lA [BM x KSTR] halves, lW [HDIM x KSTR] halves.
// Wave w owns rows [16w,16w+16); computes 8 accumulators (all 128 cols).
// Fragment layouts per CDNA5 ISA 16-bit A (16x32) / B (32x16).
// =====================================================================
template <int NK>
__device__ __forceinline__ void wmma_core(const _Float16* lA, const _Float16* lW,
                                          int wave, int lane, v8f acc[8]) {
    const int lhalf = lane >> 4;
    const int lm    = lane & 15;
#pragma unroll
    for (int c = 0; c < NK; ++c) {
        const _Float16* ap = &lA[(wave * 16 + lm) * KSTR + c * 32 + lhalf * 8];
        h8 alo = *(const h8*)ap;
        h8 ahi = *(const h8*)(ap + 16);
        v16h afrag = __builtin_shufflevector(alo, ahi,
                        0,1,2,3,4,5,6,7,8,9,10,11,12,13,14,15);
#pragma unroll
        for (int t = 0; t < 8; ++t) {
            const _Float16* bp = &lW[(t * 16 + lm) * KSTR + c * 32 + lhalf * 8];
            h8 blo = *(const h8*)bp;
            h8 bhi = *(const h8*)(bp + 16);
            v16h bfrag = __builtin_shufflevector(blo, bhi,
                            0,1,2,3,4,5,6,7,8,9,10,11,12,13,14,15);
            acc[t] = __builtin_amdgcn_wmma_f32_16x16x32_f16(
                         false, afrag, false, bfrag,
                         (short)0, acc[t], false, false);
        }
    }
}

// =====================================================================
// GEMM A: f32 input with in-register f16 conversion + K zero-padding
// (used for the narrow input projections, K=100/50). Dual f32/f16 output.
//   C = relu(A @ W^T + bias), C16 = (f16)C
// =====================================================================
template <int NK>
__global__ __launch_bounds__(256)
void k_gemm_f32in(const float* __restrict__ A, int lda, int Mrows,
                  const float* __restrict__ W, int K,
                  const float* __restrict__ bias,
                  float* __restrict__ C, _Float16* __restrict__ C16)
{
    __shared__ __attribute__((aligned(16))) _Float16 lA[BM * KSTR];
    __shared__ __attribute__((aligned(16))) _Float16 lW[HDIM * KSTR];

    const int tid  = threadIdx.x;
    const int row0 = blockIdx.x * BM;
    const int KC   = NK * 32;

    for (int i = tid; i < BM * KC; i += 256) {
        int r = i / KC, k = i - r * KC;
        int gr = row0 + r;
        float v = (gr < Mrows && k < K) ? A[gr * lda + k] : 0.0f;
        lA[r * KSTR + k] = (_Float16)v;
    }
    for (int i = tid; i < HDIM * KC; i += 256) {
        int n = i / KC, k = i - n * KC;
        lW[n * KSTR + k] = (_Float16)((k < K) ? W[n * K + k] : 0.0f);
    }
    __syncthreads();

    const int wave = tid >> 5, lane = tid & 31;
    const int lhalf = lane >> 4, lm = lane & 15;
    v8f acc[8] = {};
    wmma_core<NK>(lA, lW, wave, lane, acc);

#pragma unroll
    for (int t = 0; t < 8; ++t) {
        const int col = t * 16 + lm;
        const float badd = bias[col];
#pragma unroll
        for (int r = 0; r < 8; ++r) {
            int gr = row0 + wave * 16 + r + lhalf * 8;
            if (gr < Mrows) {
                int idx = gr * HDIM + col;
                float v = fmaxf(acc[t][r] + badd, 0.0f);
                C[idx]   = v;
                C16[idx] = (_Float16)v;
            }
        }
    }
}

// =====================================================================
// GEMM B: f16 input, K=128, tiles staged with async global->LDS copies.
//   C = rowscale*(A16 @ W16^T) + bias + (ADDC ? C : 0)
// (rowscale commutes with the GEMM, so the SAGE 1/deg mean-scaling is a
//  pure epilogue op; A16 staging is a raw byte copy -> async-LDS path.)
// =====================================================================
template <int ADDC>
__global__ __launch_bounds__(256)
void k_gemm_f16in(const _Float16* __restrict__ A16, int Mrows,
                  const _Float16* __restrict__ W16,     // [128,128] f16
                  const float* __restrict__ bias,       // may be null
                  const float* __restrict__ rowscale,   // may be null
                  float* __restrict__ C)
{
    __shared__ __attribute__((aligned(16))) _Float16 lA[BM * KSTR];
    __shared__ __attribute__((aligned(16))) _Float16 lW[HDIM * KSTR];

    const int tid  = threadIdx.x;
    const int row0 = blockIdx.x * BM;

    // stage A tile: 128 rows x 128 halves, 16B chunks, per-lane addresses
    for (int i = tid; i < BM * 16; i += 256) {
        int r = i >> 4, ck = i & 15;
        int gr = row0 + r;
        if (gr >= Mrows) gr = Mrows - 1;          // clamp: garbage rows unused
        async_copy_b128(A16 + (size_t)gr * HDIM + ck * 8, &lA[r * KSTR + ck * 8]);
    }
    // stage W tile: 128 x 128 halves
    for (int i = tid; i < HDIM * 16; i += 256) {
        int n = i >> 4, ck = i & 15;
        async_copy_b128(W16 + n * HDIM + ck * 8, &lW[n * KSTR + ck * 8]);
    }
    async_wait_all();       // this wave's copies landed in LDS
    __syncthreads();        // all waves' copies visible

    const int wave = tid >> 5, lane = tid & 31;
    const int lhalf = lane >> 4, lm = lane & 15;
    v8f acc[8] = {};
    wmma_core<4>(lA, lW, wave, lane, acc);

#pragma unroll
    for (int t = 0; t < 8; ++t) {
        const int col = t * 16 + lm;
        const float badd = bias ? bias[col] : 0.0f;
#pragma unroll
        for (int r = 0; r < 8; ++r) {
            int gr = row0 + wave * 16 + r + lhalf * 8;
            if (gr < Mrows) {
                int idx = gr * HDIM + col;
                float v = acc[t][r];
                if (rowscale) v *= rowscale[gr];
                v += badd;
                if (ADDC) v += C[idx];
                C[idx] = v;
            }
        }
    }
}

// =====================================================================
// helpers
// =====================================================================
__global__ void k_zero(float* __restrict__ p, long n) {
    long i = (long)blockIdx.x * blockDim.x + threadIdx.x;
    long s = (long)gridDim.x * blockDim.x;
    for (; i < n; i += s) p[i] = 0.0f;
}

__global__ void k_deg(const int* __restrict__ src, const int* __restrict__ dst,
                      int E, float* __restrict__ deg) {
    int e = blockIdx.x * blockDim.x + threadIdx.x;
    if (e < E) {
        unsafeAtomicAdd(&deg[dst[e]], 1.0f);
        unsafeAtomicAdd(&deg[src[e]], 1.0f);
    }
}

__global__ void k_rdeg(float* __restrict__ deg, int n) {
    int i = blockIdx.x * blockDim.x + threadIdx.x;
    if (i < n) deg[i] = 1.0f / fmaxf(deg[i], 1.0f);
}

// f32 -> f16 cast with optional per-row scale (folds 1/deg into AGG16)
__global__ void k_cast16(const float* __restrict__ src,
                         const float* __restrict__ rowscale,  // may be null
                         _Float16* __restrict__ dst, long n) {
    long stride = (long)gridDim.x * blockDim.x;
    for (long i = (long)blockIdx.x * blockDim.x + threadIdx.x; i < n; i += stride) {
        float v = src[i];
        if (rowscale) v *= rowscale[i >> 7];       // row = i / HDIM
        dst[i] = (_Float16)v;
    }
}

// undirected segment-sum, f16 gather -> f32 atomic accumulate
__global__ __launch_bounds__(256)
void k_scatter(const int* __restrict__ src, const int* __restrict__ dst, int E,
               const _Float16* __restrict__ h16, float* __restrict__ agg) {
    int e = blockIdx.x * 2 + (threadIdx.x >> 7);
    int c = threadIdx.x & (HDIM - 1);
    if (e < E) {
        int s = src[e], d = dst[e];
        float hs = (float)h16[s * HDIM + c];
        float hd = (float)h16[d * HDIM + c];
        unsafeAtomicAdd(&agg[d * HDIM + c], hs);
        unsafeAtomicAdd(&agg[s * HDIM + c], hd);
    }
}

__global__ __launch_bounds__(256)
void k_bnstats(const float* __restrict__ s, int N, float* __restrict__ sums) {
    __shared__ float ls[256], lq[256];
    int col = threadIdx.x & (HDIM - 1);
    int sub = threadIdx.x >> 7;
    float sum = 0.0f, sq = 0.0f;
    for (int r = blockIdx.x * 2 + sub; r < N; r += gridDim.x * 2) {
        float v = s[r * HDIM + col];
        sum += v; sq += v * v;
    }
    ls[threadIdx.x] = sum; lq[threadIdx.x] = sq;
    __syncthreads();
    if (threadIdx.x < HDIM) {
        sum = ls[threadIdx.x] + ls[threadIdx.x + HDIM];
        sq  = lq[threadIdx.x] + lq[threadIdx.x + HDIM];
        unsafeAtomicAdd(&sums[col], sum);
        unsafeAtomicAdd(&sums[HDIM + col], sq);
    }
}

// s = relu((s-m)*rsqrt(v+eps)*g + be) [+ residual]; optional f16 mirror
__global__ void k_bnapply(float* __restrict__ s, int N,
                          const float* __restrict__ sums,
                          const float* __restrict__ g,
                          const float* __restrict__ be,
                          const float* __restrict__ residual,   // may be null
                          _Float16* __restrict__ out16) {       // may be null
    const float invN = 1.0f / (float)N;
    int total = N * HDIM;
    int stride = gridDim.x * blockDim.x;
    for (int i = blockIdx.x * blockDim.x + threadIdx.x; i < total; i += stride) {
        int col = i & (HDIM - 1);
        float m   = sums[col] * invN;
        float var = sums[HDIM + col] * invN - m * m;
        float v = (s[i] - m) * rsqrtf(var + BN_EPS) * g[col] + be[col];
        v = fmaxf(v, 0.0f);
        if (residual) v += residual[i];
        s[i] = v;
        if (out16) out16[i] = (_Float16)v;
    }
}

// out[row] = dot(h2[row,:128], Wout) + bout ; one wave32 per row
__global__ __launch_bounds__(256)
void k_out(const float* __restrict__ h2, const float* __restrict__ Wout,
           const float* __restrict__ bout, float* __restrict__ out, int NU) {
    int row  = blockIdx.x * 8 + (threadIdx.x >> 5);
    int lane = threadIdx.x & 31;
    if (row >= NU) return;
    const float4* hp = (const float4*)(h2 + row * HDIM);
    const float4* wp = (const float4*)Wout;
    float4 a = hp[lane], w = wp[lane];
    float d = a.x * w.x + a.y * w.y + a.z * w.z + a.w * w.w;
    for (int off = 16; off > 0; off >>= 1) d += __shfl_down(d, off, 32);
    if (lane == 0) out[row] = d + bout[0];
}

// =====================================================================
// launcher
// =====================================================================
extern "C" void kernel_launch(void* const* d_in, const int* in_sizes, int n_in,
                              void* d_out, int out_size, void* d_ws, size_t ws_size,
                              hipStream_t stream) {
    const float* x_u  = (const float*)d_in[0];
    const float* x_p  = (const float*)d_in[1];
    const int*   ei   = (const int*)  d_in[2];
    const float* W_u  = (const float*)d_in[3];
    const float* b_u  = (const float*)d_in[4];
    const float* W_p  = (const float*)d_in[5];
    const float* b_p  = (const float*)d_in[6];
    const float* W1l  = (const float*)d_in[7];
    const float* b1l  = (const float*)d_in[8];
    const float* W1r  = (const float*)d_in[9];
    const float* g1   = (const float*)d_in[10];
    const float* be1  = (const float*)d_in[11];
    const float* W2l  = (const float*)d_in[12];
    const float* b2l  = (const float*)d_in[13];
    const float* W2r  = (const float*)d_in[14];
    const float* g2   = (const float*)d_in[15];
    const float* be2  = (const float*)d_in[16];
    const float* Wout = (const float*)d_in[17];
    const float* bout = (const float*)d_in[18];
    float* out = (float*)d_out;

    const int* src = ei;
    const int* dst = ei + EED_;

    // workspace: X | AGG | S | DEG | SUMS | X16 | S16 | AGG16 | W16[4]
    const size_t NH = (size_t)NTOT * HDIM;
    float* X    = (float*)d_ws;
    float* AGG  = X + NH;
    float* S    = AGG + NH;
    float* DEG  = S + NH;
    float* SUMS = DEG + NTOT;
    _Float16* X16   = (_Float16*)(SUMS + 2 * HDIM);
    _Float16* S16   = X16 + NH;
    _Float16* AGG16 = S16 + NH;
    _Float16* W16_1l = AGG16 + NH;
    _Float16* W16_1r = W16_1l + HDIM * HDIM;
    _Float16* W16_2l = W16_1r + HDIM * HDIM;
    _Float16* W16_2r = W16_2l + HDIM * HDIM;

    const int gN  = (NTOT + BM - 1) / BM;
    const int gNU = (NU_  + BM - 1) / BM;
    const int gNP = (NP_  + BM - 1) / BM;
    const long WN = (long)HDIM * HDIM;

    // ---- degrees + zero init ----
    k_zero<<<256, 256, 0, stream>>>(DEG, (long)NTOT + 2 * HDIM);
    k_zero<<<2048, 256, 0, stream>>>(AGG, (long)NH);
    k_deg<<<(EED_ + 255) / 256, 256, 0, stream>>>(src, dst, EED_, DEG);
    k_rdeg<<<(NTOT + 255) / 256, 256, 0, stream>>>(DEG, NTOT);

    // ---- f16 weight mirrors for the H x H GEMMs ----
    k_cast16<<<16, 256, 0, stream>>>(W1l, nullptr, W16_1l, WN);
    k_cast16<<<16, 256, 0, stream>>>(W1r, nullptr, W16_1r, WN);
    k_cast16<<<16, 256, 0, stream>>>(W2l, nullptr, W16_2l, WN);
    k_cast16<<<16, 256, 0, stream>>>(W2r, nullptr, W16_2r, WN);

    // ---- input projections (ReLU) -> X (f32) + X16 (f16) ----
    k_gemm_f32in<4><<<gNU, 256, 0, stream>>>(x_u, DU_, NU_, W_u, DU_, b_u, X, X16);
    k_gemm_f32in<2><<<gNP, 256, 0, stream>>>(x_p, DP_, NP_, W_p, DP_, b_p,
                                             X + (size_t)NU_ * HDIM, X16 + (size_t)NU_ * HDIM);

    // ---- SAGE layer 1 ----
    k_scatter<<<(EED_ + 1) / 2, 256, 0, stream>>>(src, dst, EED_, X16, AGG);
    k_cast16<<<2048, 256, 0, stream>>>(AGG, DEG, AGG16, (long)NH);   // mean in f16
    // S = AGG16 @ W1l^T + b1l   (rowscale folded into AGG16 already)
    k_gemm_f16in<0><<<gN, 256, 0, stream>>>(AGG16, NTOT, W16_1l, b1l, nullptr, S);
    // S += X16 @ W1r^T
    k_gemm_f16in<1><<<gN, 256, 0, stream>>>(X16, NTOT, W16_1r, nullptr, nullptr, S);
    k_bnstats<<<512, 256, 0, stream>>>(S, NTOT, SUMS);
    k_bnapply<<<2048, 256, 0, stream>>>(S, NTOT, SUMS, g1, be1, nullptr, S16); // S=h1, S16=h1(f16)

    // ---- SAGE layer 2 ----
    k_zero<<<2048, 256, 0, stream>>>(AGG, (long)NH);
    k_zero<<<1, 256, 0, stream>>>(SUMS, 2 * HDIM);
    k_scatter<<<(EED_ + 1) / 2, 256, 0, stream>>>(src, dst, EED_, S16, AGG);
    k_cast16<<<2048, 256, 0, stream>>>(AGG, DEG, AGG16, (long)NH);
    // AGG(f32) = AGG16 @ W2l^T + b2l   (AGG f32 is dead after the cast -> reuse as C)
    k_gemm_f16in<0><<<gN, 256, 0, stream>>>(AGG16, NTOT, W16_2l, b2l, nullptr, AGG);
    // AGG += S16 @ W2r^T
    k_gemm_f16in<1><<<gN, 256, 0, stream>>>(S16, NTOT, W16_2r, nullptr, nullptr, AGG);
    k_bnstats<<<512, 256, 0, stream>>>(AGG, NTOT, SUMS);
    k_bnapply<<<2048, 256, 0, stream>>>(AGG, NTOT, SUMS, g2, be2, X, nullptr); // h2

    // ---- output head ----
    k_out<<<(NU_ + 7) / 8, 256, 0, stream>>>(AGG, Wout, bout, out, NU_);
}
